// LeakyRNN_22007412425254
// MI455X (gfx1250) — compile-verified, matrix operand order checked by reading, not run on hardware
//
#include <hip/hip_runtime.h>
#include <hip/hip_bf16.h>

// ---------------------------------------------------------------------------
// Leaky RNN on MI455X (gfx1250).
// Recurrent GEMM (64x2048x2048 per step, 200 steps) on the WMMA pipe:
// v_wmma_f32_16x16x32_bf16, f32 accumulate, f32 state.
// The GEMM is L2-traffic bound -> each wave register-blocks a 2x2 group of
// 16x16 output tiles (4 WMMAs per 4 fragment loads), and K is split across
// 2 waves per block with an LDS combine to keep 256 waves in flight.
// J is converted once to bf16 pre-swizzled into the WMMA B-fragment layout;
// tanh(h) is packed per-step into the A-fragment layout.
// ---------------------------------------------------------------------------

typedef __attribute__((ext_vector_type(16))) __bf16 v16bf;
typedef __attribute__((ext_vector_type(8)))  float  v8f;

#define NN      2048      // neurons
#define BB      64        // batch
#define TT      200       // timesteps
#define NI      32        // inputs
#define NO      8         // outputs
#define ALPHA   0.2f      // DT/TAU
#define INV_N   (1.0f/2048.0f)

// bf16 round-to-nearest-even from f32 (bit manipulation; inputs are finite)
__device__ __forceinline__ unsigned int f2bf_bits(float x) {
    unsigned int u = __builtin_bit_cast(unsigned int, x);
    return (u + 0x7FFFu + ((u >> 16) & 1u)) >> 16;
}

__device__ __forceinline__ v8f wmma_bf16(v16bf a, v16bf b, v8f c) {
    return __builtin_amdgcn_wmma_f32_16x16x32_bf16(
        /*neg_a=*/false, a, /*neg_b=*/false, b,
        /*c_mod=*/(short)0, c, /*reuse_a=*/false, /*reuse_b=*/false);
}

// ---------------------------------------------------------------------------
// Fragment layout (CDNA5 ISA 7.12.2, 16-bit A 16x32; B mirrored).
// Packed dword index for tile-local (lane, v):  tileBase*256 + lane*8 + v
//   A tile (m x k):  m = lane&15,  k = (v>>2)*16 + (lane>>4)*8 + (v&3)*2 (+0/1)
//   B tile (k x n):  n = lane&15,  k = same formula
// ---------------------------------------------------------------------------

// One-time: J[n][k] (f32 row-major) -> Jpack bf16 B-fragments, nt-major /
// kt-minor so each output column tile's K stream is a contiguous 64KB block.
__global__ __launch_bounds__(256) void pack_J_bf16(
    const float* __restrict__ J, unsigned int* __restrict__ Jpack)
{
    unsigned int g  = blockIdx.x * 256u + threadIdx.x;   // 0 .. 2^21-1
    unsigned int v  = g & 7u;
    unsigned int l  = (g >> 3) & 31u;
    unsigned int kt = (g >> 8) & 63u;
    unsigned int nt = g >> 14;                           // 0..127
    unsigned int n  = nt * 16u + (l & 15u);
    unsigned int k  = kt * 32u + (v >> 2) * 16u + (l >> 4) * 8u + (v & 3u) * 2u;
    float x0 = J[n * NN + k];
    float x1 = J[n * NN + k + 1];
    Jpack[g] = f2bf_bits(x0) | (f2bf_bits(x1) << 16);
}

// h_cur <- broadcast(h0)
__global__ __launch_bounds__(256) void init_h(
    const float* __restrict__ h0, float* __restrict__ h)
{
    int e = blockIdx.x * 256 + threadIdx.x;              // < 131072
    h[e] = h0[e & (NN - 1)];
}

// Per-step prep (all phases depend only on h_t; mutually independent):
//  A) tanh(h) -> bf16 A-fragments          (if do_step)
//  B) dn = u_t @ In_w.T + noise_t  (f32)   (if do_step)
//  C) y[:, t, :] = Out_w.T @ tanh(h) / N   (always; blocks 0..63)
__global__ __launch_bounds__(256) void rnn_prep(
    const float* __restrict__ h,        // [64,2048]
    const float* __restrict__ u,        // [64,200,32]
    const float* __restrict__ In_w,     // [2048,32]
    const float* __restrict__ Out_w,    // [2048,8]
    const float* __restrict__ noise,    // [64,200,2048]
    unsigned int* __restrict__ Apack,   // 65536 dwords
    float* __restrict__ dn,             // [64,2048]
    float* __restrict__ y,              // [64,201,8]
    int t, int do_step)
{
    const int tid = threadIdx.x;
    const unsigned int g = blockIdx.x * 256u + tid;      // grid = 256x256 = 65536

    if (do_step) {
        // --- Phase A: tanh + pack into WMMA A-fragment layout --------------
        {
            unsigned int v  = g & 7u;
            unsigned int l  = (g >> 3) & 31u;
            unsigned int kt = (g >> 8) & 63u;
            unsigned int mt = (g >> 14) & 3u;
            unsigned int b  = mt * 16u + (l & 15u);
            unsigned int k  = kt * 32u + (v >> 2) * 16u + (l >> 4) * 8u + (v & 3u) * 2u;
            float x0 = tanhf(h[b * NN + k]);
            float x1 = tanhf(h[b * NN + k + 1]);
            Apack[g] = f2bf_bits(x0) | (f2bf_bits(x1) << 16);
        }
        // --- Phase B: drive + noise ----------------------------------------
        for (unsigned int e = g; e < (unsigned)(BB * NN); e += 65536u) {
            unsigned int b = e >> 11;
            unsigned int n = e & (NN - 1);
            const float* up = u + ((size_t)b * TT + t) * NI;
            const float* wp = In_w + (size_t)n * NI;
            float acc = 0.f;
            #pragma unroll
            for (int i = 0; i < NI; ++i) acc = fmaf(up[i], wp[i], acc);
            dn[e] = acc + noise[((size_t)b * TT + t) * NN + n];
        }
    }

    // --- Phase C: output projection (block-uniform branch) ------------------
    if (blockIdx.x < BB) {
        __shared__ float sdata[256 * NO];
        const int b = blockIdx.x;
        float acc[NO];
        #pragma unroll
        for (int o = 0; o < NO; ++o) acc[o] = 0.f;
        for (int n = tid; n < NN; n += 256) {
            float a = tanhf(h[b * NN + n]);
            const float* op = Out_w + (size_t)n * NO;
            #pragma unroll
            for (int o = 0; o < NO; ++o) acc[o] = fmaf(op[o], a, acc[o]);
        }
        #pragma unroll
        for (int o = 0; o < NO; ++o) sdata[tid * NO + o] = acc[o];
        __syncthreads();
        if (tid < NO) {
            float s = 0.f;
            for (int j = 0; j < 256; ++j) s += sdata[j * NO + tid];
            y[((size_t)b * (TT + 1) + t) * NO + tid] = s * INV_N;
        }
    }
}

// One recurrence step on the WMMA pipe.
// Grid: 128 blocks x 64 threads (2 waves).
//   block -> 2x2 group of 16x16 tiles: mg in [0,2) picks batch tiles
//            {2mg, 2mg+1}, ng in [0,64) picks column tiles {2ng, 2ng+1}.
//   wave w handles K-half w (kt 32w .. 32w+31); partials combined via LDS.
// Inner loop: 4 fragment loads feed 4 WMMAs (register-level A/B reuse).
__global__ __launch_bounds__(64) void rnn_step_wmma(
    const unsigned int* __restrict__ Apack,
    const unsigned int* __restrict__ Jpack,
    const float* __restrict__ h_cur,
    const float* __restrict__ dn,
    float* __restrict__ h_next)
{
    const int lane = threadIdx.x & 31;
    const int wv   = threadIdx.x >> 5;     // K-half: 0 or 1
    const int mg   = blockIdx.x >> 6;      // 0..1
    const int ng   = blockIdx.x & 63;      // 0..63

    const v16bf* __restrict__ Ap = (const v16bf*)Apack;
    const v16bf* __restrict__ Bp = (const v16bf*)Jpack;
    const v16bf* a0 = Ap + (size_t)((mg * 2 + 0) * 64) * 32 + lane;
    const v16bf* a1 = Ap + (size_t)((mg * 2 + 1) * 64) * 32 + lane;
    const v16bf* b0 = Bp + (size_t)((ng * 2 + 0) * 64) * 32 + lane;
    const v16bf* b1 = Bp + (size_t)((ng * 2 + 1) * 64) * 32 + lane;

    v8f c00 = {}, c01 = {}, c10 = {}, c11 = {};
    const int kt0 = wv * 32;
    #pragma unroll 2
    for (int kt = kt0; kt < kt0 + 32; ++kt) {
        v16bf A0 = a0[kt * 32];
        v16bf A1 = a1[kt * 32];
        v16bf B0 = b0[kt * 32];
        v16bf B1 = b1[kt * 32];
        c00 = wmma_bf16(A0, B0, c00);
        c01 = wmma_bf16(A0, B1, c01);
        c10 = wmma_bf16(A1, B0, c10);
        c11 = wmma_bf16(A1, B1, c11);
    }

    // Combine K-halves: wave 1 stages partials in LDS, wave 0 accumulates.
    __shared__ float red[4 * 8 * 32];      // 4 KB
    if (wv == 1) {
        #pragma unroll
        for (int v = 0; v < 8; ++v) {
            red[(0 * 8 + v) * 32 + lane] = c00[v];
            red[(1 * 8 + v) * 32 + lane] = c01[v];
            red[(2 * 8 + v) * 32 + lane] = c10[v];
            red[(3 * 8 + v) * 32 + lane] = c11[v];
        }
    }
    __syncthreads();
    if (wv == 0) {
        const int col = lane & 15;
        const int rh  = (lane >> 4) * 8;
        #pragma unroll
        for (int i = 0; i < 2; ++i) {
            #pragma unroll
            for (int j = 0; j < 2; ++j) {
                const int tile = i * 2 + j;
                const int mt = mg * 2 + i;
                const int nt = ng * 2 + j;
                #pragma unroll
                for (int v = 0; v < 8; ++v) {
                    float cv;
                    if (tile == 0)      cv = c00[v];
                    else if (tile == 1) cv = c01[v];
                    else if (tile == 2) cv = c10[v];
                    else                cv = c11[v];
                    cv += red[(tile * 8 + v) * 32 + lane];
                    const int m = mt * 16 + rh + v;
                    const int n = nt * 16 + col;
                    const size_t idx = (size_t)m * NN + n;
                    h_next[idx] = (1.0f - ALPHA) * h_cur[idx]
                                + ALPHA * (cv * INV_N + dn[idx]);
                }
            }
        }
    }
}

// ---------------------------------------------------------------------------
extern "C" void kernel_launch(void* const* d_in, const int* in_sizes, int n_in,
                              void* d_out, int out_size, void* d_ws, size_t ws_size,
                              hipStream_t stream)
{
    const float* u     = (const float*)d_in[0];  // [64,200,32]
    const float* In_w  = (const float*)d_in[1];  // [2048,32]
    const float* Out_w = (const float*)d_in[2];  // [2048,8]
    const float* J     = (const float*)d_in[3];  // [2048,2048]
    const float* h0    = (const float*)d_in[4];  // [2048]
    const float* noise = (const float*)d_in[5];  // [64,200,2048]
    float* y = (float*)d_out;                    // [64,201,8] f32

    // Workspace layout (~9.75 MB total)
    uint8_t* ws = (uint8_t*)d_ws;
    unsigned int* Jpack = (unsigned int*)(ws);                          // 8 MB
    unsigned int* Apack = (unsigned int*)(ws + (8u << 20));             // 256 KB
    float* dn = (float*)(ws + (8u << 20) + (256u << 10));               // 512 KB
    float* hA = (float*)(ws + (8u << 20) + (768u << 10));               // 512 KB
    float* hB = (float*)(ws + (8u << 20) + (1280u << 10));              // 512 KB

    // One-time: J -> bf16 B-fragments (L2-resident thereafter)
    pack_J_bf16<<<8192, 256, 0, stream>>>(J, Jpack);
    // h_0 = broadcast(h0)
    init_h<<<512, 256, 0, stream>>>(h0, hA);

    float* hc = hA;
    float* hn = hB;
    for (int t = 0; t < TT; ++t) {
        rnn_prep<<<256, 256, 0, stream>>>(hc, u, In_w, Out_w, noise,
                                          Apack, dn, y, t, /*do_step=*/1);
        rnn_step_wmma<<<128, 64, 0, stream>>>(Apack, Jpack, hc, dn, hn);
        float* tmp = hc; hc = hn; hn = tmp;
    }
    // Final output row y[:, 200, :]
    rnn_prep<<<256, 256, 0, stream>>>(hc, u, In_w, Out_w, noise,
                                      Apack, dn, y, TT, /*do_step=*/0);
}
